// GATv2Layer_38397007626741
// MI455X (gfx1250) — matrix operand order repaired and checked by reading, not compiled, not run on hardware
//
#include <hip/hip_runtime.h>
#include <hip/hip_bf16.h>

// ---------------------------------------------------------------------------
// GATv2 on kNN graph, CDNA5 (gfx1250, wave32, WMMA + async LDS copies).
// Pipeline:
//   1) k_split_norm : x -> bf16 hi/lo split + row squared norms
//   2) k_splitW     : Wl,Wr -> transposed bf16 hi/lo splits (B-operand layout)
//   3) k_gemm_xw    : xl = x@Wl, xr = x@Wr via v_wmma_f32_16x16x32_bf16 (3-prod)
//   4) k_knn        : candidate chunks async-staged to LDS (double buffered,
//                     shared by 8 waves/block), Gram tiles via WMMA, per-target
//                     top-16 kept in registers
//   5) k_attend     : gather, leaky-relu, attention softmax(k=16), head mean
// ---------------------------------------------------------------------------

#define NN      16384
#define DD      128
#define HH      3
#define CC      128
#define HC      384
#define KNB     16          // neighbors
#define NEG_SLOPE 0.2f

typedef __attribute__((ext_vector_type(16))) __bf16 v16bf;
typedef __attribute__((ext_vector_type(8)))  __bf16 v8bf;
typedef __attribute__((ext_vector_type(8)))  float  v8f;

__device__ __forceinline__ v8f wmma_bf16(v16bf a, v16bf b, v8f c) {
    // 8 args: (neg_a, A, neg_b, B, c_mod, C, reuse_a, reuse_b)
    return __builtin_amdgcn_wmma_f32_16x16x32_bf16(
        false, a, false, b, (short)0, c, false, false);
}

// A-operand (16x32 bf16, row-major source): lane holds row m=lane&15,
// half = lane>>4; needs K chunk [ks+half*16,+16) (contiguous, matches B).
__device__ __forceinline__ v16bf load_a_pair(const __bf16* p) {
    v8bf lo = *(const v8bf*)(p);
    v8bf hi = *(const v8bf*)(p + 8);
    return __builtin_shufflevector(lo, hi,
        0,1,2,3,4,5,6,7,8,9,10,11,12,13,14,15);
}

// ---------------------------------------------------------------------------
// 1) split x into bf16 hi/lo + row norms.  grid = N, block = 128
// ---------------------------------------------------------------------------
__global__ void __launch_bounds__(128)
k_split_norm(const float* __restrict__ x, __bf16* __restrict__ xhi,
             __bf16* __restrict__ xlo, float* __restrict__ sq)
{
    const int i = blockIdx.x;
    const int c = threadIdx.x;
    float v = x[i * DD + c];
    __bf16 h = (__bf16)v;
    float rem = v - (float)h;
    xhi[i * DD + c] = h;
    xlo[i * DD + c] = (__bf16)rem;

    float s = v * v;
    #pragma unroll
    for (int off = 16; off; off >>= 1) s += __shfl_xor(s, off, 32);
    __shared__ float part[4];
    if ((c & 31) == 0) part[c >> 5] = s;
    __syncthreads();
    if (c == 0) sq[i] = part[0] + part[1] + part[2] + part[3];
}

// ---------------------------------------------------------------------------
// 2) transpose+split Wl/Wr [K=128 x 384] -> Wt [384 x 128] bf16 hi/lo
// ---------------------------------------------------------------------------
__global__ void __launch_bounds__(256)
k_splitW(const float* __restrict__ Wl, const float* __restrict__ Wr,
         __bf16* __restrict__ wlth, __bf16* __restrict__ wltl,
         __bf16* __restrict__ wrth, __bf16* __restrict__ wrtl)
{
    int t = blockIdx.x * blockDim.x + threadIdx.x;
    const int per = HC * DD;            // 49152
    if (t >= 2 * per) return;
    int mat = t / per;
    int r   = t % per;
    int n   = r / DD;
    int kk  = r % DD;
    float v = (mat ? Wr : Wl)[kk * HC + n];
    __bf16 h = (__bf16)v;
    __bf16 l = (__bf16)(v - (float)h);
    if (mat == 0) { wlth[r] = h; wltl[r] = l; }
    else          { wrth[r] = h; wrtl[r] = l; }
}

// ---------------------------------------------------------------------------
// 3) xl/xr GEMM: [16384x128] @ [128x384] via WMMA (split-bf16, 3 products).
//    One wave per 16x16 output tile. grid = 3072, block = 256 (8 waves).
// ---------------------------------------------------------------------------
__global__ void __launch_bounds__(256)
k_gemm_xw(const __bf16* __restrict__ xhi, const __bf16* __restrict__ xlo,
          const __bf16* __restrict__ wth, const __bf16* __restrict__ wtl,
          float* __restrict__ out)
{
    const int wave = threadIdx.x >> 5;
    const int lane = threadIdx.x & 31;
    const int lm = lane & 15, hl = lane >> 4;
    const int tile  = blockIdx.x * 8 + wave;
    const int rt = tile / (HC / 16), ct = tile % (HC / 16);
    const int ibase = rt * 16, nbase = ct * 16;

    v16bf Ahi[4], Alo[4];
    const int arow = (ibase + lm) * DD;
    #pragma unroll
    for (int ks = 0; ks < 4; ++ks) {
        int off = arow + ks * 32 + hl * 16;
        Ahi[ks] = load_a_pair(xhi + off);
        Alo[ks] = load_a_pair(xlo + off);
    }
    v8f acc = {0.f,0.f,0.f,0.f,0.f,0.f,0.f,0.f};
    const int brow = (nbase + lm) * DD;
    #pragma unroll
    for (int ks = 0; ks < 4; ++ks) {
        int off = brow + ks * 32 + hl * 16;       // 32B aligned
        v16bf Bhi = *(const v16bf*)(wth + off);
        v16bf Blo = *(const v16bf*)(wtl + off);
        acc = wmma_bf16(Ahi[ks], Bhi, acc);
        acc = wmma_bf16(Ahi[ks], Blo, acc);
        acc = wmma_bf16(Alo[ks], Bhi, acc);
    }
    #pragma unroll
    for (int v = 0; v < 8; ++v) {
        int m = v + hl * 8;
        out[(ibase + m) * HC + nbase + lm] = acc[v];
    }
}

// ---------------------------------------------------------------------------
// 4) fused kNN. Block = 8 waves = 128 target rows. Candidate chunks (32 rows
//    of bf16 hi/lo = 16KB) are async-copied into a double-buffered LDS stage
//    shared by all 8 waves (8x L2 traffic reduction), overlapped with WMMA.
//    Per-lane top-16 half-lists live in registers; merged at the end through
//    the (then-dead) stage buffer.
// ---------------------------------------------------------------------------
__global__ void __launch_bounds__(256)
k_knn(const __bf16* __restrict__ xhi, const __bf16* __restrict__ xlo,
      const float* __restrict__ sq, int* __restrict__ knn)
{
    __shared__ __align__(16) __bf16 s_stage[2][2][32][DD];   // [buf][hi/lo] 32 KB
    __shared__ float s_dot[8][2][16][16];                    // 16 KB
    __shared__ float s_sqj[8][32];

    const int w    = threadIdx.x >> 5;
    const int lane = threadIdx.x & 31;
    const int lm = lane & 15, hl = lane >> 4;
    const int ibase = (blockIdx.x * 8 + w) * 16;

    // A operand: 16 target rows, full K=128, hi/lo (register resident)
    v16bf Ahi[4], Alo[4];
    const int arow = (ibase + lm) * DD;
    #pragma unroll
    for (int ks = 0; ks < 4; ++ks) {
        int off = arow + ks * 32 + hl * 16;
        Ahi[ks] = load_a_pair(xhi + off);
        Alo[ks] = load_a_pair(xlo + off);
    }

    // register-resident top-16 half-list (target lm, candidate-part hl)
    float listd[KNB];
    int   listi[KNB];
    #pragma unroll
    for (int s = 0; s < KNB; ++s) { listd[s] = __builtin_inff(); listi[s] = -1; }
    float worstv = __builtin_inff();
    int   worsts = 0;
    const int   mytar = ibase + lm;
    const float sqi   = sq[mytar];

    // async stage: this wave copies its 1KB slice of each 8KB stream chunk.
    // the imm offset is added to both LDS and global addresses, so one
    // instruction moves 32 lanes x 16B; offset:512 covers the second half.
    const unsigned lds_hi0 = (unsigned)(unsigned long long)(&s_stage[0][0][0][0]);
    const unsigned lds_lo0 = (unsigned)(unsigned long long)(&s_stage[0][1][0][0]);
    const unsigned bufstep = (unsigned)sizeof(s_stage[0]);   // 16384 bytes
    const unsigned loff    = (unsigned)(w * 1024 + lane * 16);

    #define STAGE_CHUNK(cb_, buf_)                                              \
    do {                                                                        \
        unsigned long long gb = (unsigned long long)(cb_) * (32u * DD * 2u);    \
        unsigned long long ghi = (unsigned long long)xhi + gb + loff;           \
        unsigned long long glo = (unsigned long long)xlo + gb + loff;           \
        unsigned dhi = lds_hi0 + (unsigned)(buf_) * bufstep + loff;             \
        unsigned dlo = lds_lo0 + (unsigned)(buf_) * bufstep + loff;             \
        asm volatile("global_load_async_to_lds_b128 %0, %1, off"                \
                     :: "v"(dhi), "v"(ghi) : "memory");                         \
        asm volatile("global_load_async_to_lds_b128 %0, %1, off offset:512"    \
                     :: "v"(dhi), "v"(ghi) : "memory");                         \
        asm volatile("global_load_async_to_lds_b128 %0, %1, off"                \
                     :: "v"(dlo), "v"(glo) : "memory");                         \
        asm volatile("global_load_async_to_lds_b128 %0, %1, off offset:512"    \
                     :: "v"(dlo), "v"(glo) : "memory");                         \
    } while (0)

    STAGE_CHUNK(0, 0);
    asm volatile("s_wait_asynccnt 0x0" ::: "memory");
    __syncthreads();

    for (int cb = 0; cb < NN / 32; ++cb) {
        const int buf = cb & 1;
        if (cb + 1 < NN / 32) STAGE_CHUNK(cb + 1, buf ^ 1);

        const int jbase = cb * 32;
        s_sqj[w][lane] = sq[jbase + lane];

        #pragma unroll
        for (int tl = 0; tl < 2; ++tl) {
            v8f acc = {0.f,0.f,0.f,0.f,0.f,0.f,0.f,0.f};
            const __bf16* bh = &s_stage[buf][0][tl * 16 + lm][0];
            const __bf16* bl = &s_stage[buf][1][tl * 16 + lm][0];
            #pragma unroll
            for (int ks = 0; ks < 4; ++ks) {
                int off = ks * 32 + hl * 16;            // 32B aligned in LDS
                v16bf Bhi = *(const v16bf*)(bh + off);
                v16bf Blo = *(const v16bf*)(bl + off);
                acc = wmma_bf16(Ahi[ks], Bhi, acc);
                acc = wmma_bf16(Ahi[ks], Blo, acc);
                acc = wmma_bf16(Alo[ks], Bhi, acc);
            }
            #pragma unroll
            for (int v = 0; v < 8; ++v)
                s_dot[w][tl][v + hl * 8][lm] = acc[v];
        }
        asm volatile("s_wait_dscnt 0x0" ::: "memory");   // wave-internal handoff

        // selection: lane scans its 16 candidates (part hl) for target lm
        #pragma unroll 1
        for (int n = 0; n < 16; ++n) {
            int j = jbase + hl * 16 + n;
            float dot  = s_dot[w][hl][lm][n];
            float dist = sqi + s_sqj[w][hl * 16 + n] - 2.0f * dot;
            if (j != mytar && dist < worstv) {
                #pragma unroll
                for (int s = 0; s < KNB; ++s)
                    if (s == worsts) { listd[s] = dist; listi[s] = j; }
                float wv = -__builtin_inff(); int wsl = 0;
                #pragma unroll
                for (int s = 0; s < KNB; ++s)
                    if (listd[s] > wv) { wv = listd[s]; wsl = s; }
                worstv = wv; worsts = wsl;
            }
        }

        asm volatile("s_wait_asynccnt 0x0" ::: "memory");
        __syncthreads();   // next buffer staged & everyone done with this one
    }

    // merge the two half-lists per target through the now-dead stage buffer.
    float* md = (float*)&s_stage[0][0][0][0];   // 16 KB: 8*32*16 floats
    int*   mi = (int*)  &s_stage[1][0][0][0];   // 16 KB: 8*32*16 ints
    const int mybase = (w * 32 + lane) * KNB;
    #pragma unroll
    for (int s = 0; s < KNB; ++s) { md[mybase + s] = listd[s]; mi[mybase + s] = listi[s]; }
    asm volatile("s_wait_dscnt 0x0" ::: "memory");

    if (hl == 0) {
        const int b0 = (w * 32 + lm) * KNB;
        const int b1 = (w * 32 + lm + 16) * KNB;
        #pragma unroll 1
        for (int r = 0; r < KNB; ++r) {
            float best = __builtin_inff(); int bp = 0, bs = 0;
            #pragma unroll
            for (int s = 0; s < KNB; ++s) {
                float d0 = md[b0 + s];
                if (d0 < best) { best = d0; bp = 0; bs = s; }
                float d1 = md[b1 + s];
                if (d1 < best) { best = d1; bp = 1; bs = s; }
            }
            const int bb = bp ? b1 : b0;
            knn[mytar * KNB + r] = mi[bb + bs];
            md[bb + bs] = __builtin_inff();
        }
    }
    #undef STAGE_CHUNK
}

// ---------------------------------------------------------------------------
// 5) attention epilogue: one block (128 threads = channels) per node.
// ---------------------------------------------------------------------------
__global__ void __launch_bounds__(128)
k_attend(const float* __restrict__ xl, const float* __restrict__ xr,
         const int* __restrict__ knn, const float* __restrict__ att,
         const float* __restrict__ bias, float* __restrict__ out)
{
    const int i = blockIdx.x;
    const int c = threadIdx.x;

    __shared__ int   sidx[KNB];
    __shared__ float sxl[KNB][HH][CC];
    __shared__ float spart[HH][4];
    __shared__ float se[KNB][HH];
    __shared__ float salpha[KNB][HH];

    if (c < KNB) sidx[c] = knn[i * KNB + c];
    __syncthreads();

    float xrv[HH], attv[HH];
    #pragma unroll
    for (int h = 0; h < HH; ++h) {
        xrv[h]  = xr[i * HC + h * CC + c];
        attv[h] = att[h * CC + c];
    }
    const int wlane = c & 31, wid = c >> 5;

    for (int kk = 0; kk < KNB; ++kk) {
        const int j = sidx[kk];
        float pv[HH];
        #pragma unroll
        for (int h = 0; h < HH; ++h) {
            float xv = xl[j * HC + h * CC + c];
            sxl[kk][h][c] = xv;
            float g = xv + xrv[h];
            g = g > 0.0f ? g : NEG_SLOPE * g;
            pv[h] = attv[h] * g;
        }
        #pragma unroll
        for (int h = 0; h < HH; ++h) {
            float s = pv[h];
            #pragma unroll
            for (int off = 16; off; off >>= 1) s += __shfl_xor(s, off, 32);
            if (wlane == 0) spart[h][wid] = s;
        }
        __syncthreads();
        if (c < HH) se[kk][c] = spart[c][0] + spart[c][1] + spart[c][2] + spart[c][3];
        __syncthreads();
    }

    if (c < HH) {   // softmax over k per head
        float mx = -__builtin_inff();
        for (int kk = 0; kk < KNB; ++kk) mx = fmaxf(mx, se[kk][c]);
        float ssum = 0.0f;
        for (int kk = 0; kk < KNB; ++kk) {
            float ev = __expf(se[kk][c] - mx);
            salpha[kk][c] = ev; ssum += ev;
        }
        float inv = 1.0f / ssum;
        for (int kk = 0; kk < KNB; ++kk) salpha[kk][c] *= inv;
    }
    __syncthreads();

    float acc = 0.0f;
    for (int kk = 0; kk < KNB; ++kk) {
        #pragma unroll
        for (int h = 0; h < HH; ++h)
            acc += salpha[kk][h] * sxl[kk][h][c];
    }
    out[i * CC + c] = acc * (1.0f / (float)HH) + bias[c];
}

// ---------------------------------------------------------------------------
// launch
// ---------------------------------------------------------------------------
extern "C" void kernel_launch(void* const* d_in, const int* in_sizes, int n_in,
                              void* d_out, int out_size, void* d_ws, size_t ws_size,
                              hipStream_t stream)
{
    (void)in_sizes; (void)n_in; (void)out_size; (void)ws_size;

    const float* x    = (const float*)d_in[0];
    const float* Wl   = (const float*)d_in[1];
    const float* Wr   = (const float*)d_in[2];
    const float* att  = (const float*)d_in[3];
    const float* bias = (const float*)d_in[4];
    // d_in[5] == k, fixed at 16 by the reference setup (hard-coded: reading a
    // device scalar on host would break graph capture).

    char* ws = (char*)d_ws;
    size_t off = 0;
    auto take = [&](size_t bytes) -> void* {
        void* p = ws + off;
        off += (bytes + 255) & ~(size_t)255;
        return p;
    };
    __bf16* xhi  = (__bf16*)take((size_t)NN * DD * 2);
    __bf16* xlo  = (__bf16*)take((size_t)NN * DD * 2);
    float*  sq   = (float*) take((size_t)NN * 4);
    __bf16* wlth = (__bf16*)take((size_t)HC * DD * 2);
    __bf16* wltl = (__bf16*)take((size_t)HC * DD * 2);
    __bf16* wrth = (__bf16*)take((size_t)HC * DD * 2);
    __bf16* wrtl = (__bf16*)take((size_t)HC * DD * 2);
    float*  xlb  = (float*) take((size_t)NN * HC * 4);
    float*  xrb  = (float*) take((size_t)NN * HC * 4);
    int*    idx  = (int*)   take((size_t)NN * KNB * 4);

    k_split_norm<<<NN, 128, 0, stream>>>(x, xhi, xlo, sq);
    k_splitW<<<(2 * HC * DD + 255) / 256, 256, 0, stream>>>(Wl, Wr, wlth, wltl, wrth, wrtl);

    const int tiles = (NN / 16) * (HC / 16);       // 24576
    k_gemm_xw<<<tiles / 8, 256, 0, stream>>>(xhi, xlo, wlth, wltl, xlb);
    k_gemm_xw<<<tiles / 8, 256, 0, stream>>>(xhi, xlo, wrth, wrtl, xrb);

    k_knn<<<(NN / 16) / 8, 256, 0, stream>>>(xhi, xlo, sq, idx);

    k_attend<<<NN, 128, 0, stream>>>(xlb, xrb, idx, att, bias, (float*)d_out);
}
